// FastDAGGRU_6305011990816
// MI455X (gfx1250) — compile-verified, matrix operand order checked by reading, not run on hardware
//
#include <hip/hip_runtime.h>
#include <hip/hip_bf16.h>

typedef __attribute__((ext_vector_type(16))) __bf16 v16bf;
typedef __attribute__((ext_vector_type(8)))  float  v8f;

#define NTOT   200000
#define NPL    12500
#define NLVL   16
#define DEG    8
#define EPL    (NPL * DEG)      // 100000 edges per level
#define HD     128
#define G3     384              // 3*H gate columns
#define WMAT   (24 * 4 * 32 * 16)  // one swizzled 128x384 matrix = 49152 bf16

static __device__ __forceinline__ __bf16 f2bf(float f) {
    union { float f; unsigned u; } x; x.f = f;
    unsigned r = (x.u + 0x7FFFu + ((x.u >> 16) & 1u)) >> 16;   // RNE
    union { unsigned short s; __bf16 b; } y; y.s = (unsigned short)r;
    return y.b;
}

// Fragment index helper: element e (0..15) of lane's A/B fragment for one
// 16x16x32 bf16 WMMA K-window. Returns k offset within the 32-wide window.
static __device__ __forceinline__ int frag_k(int lane, int e) {
    int j = e >> 1, p = e & 1;
    return ((j >> 2) << 4) + ((lane >> 4) << 3) + ((j & 3) << 1) + p;
}

// ---------------------------------------------------------------------------
// Pre-swizzle Wx (2 layers x 128x384) and Wh (2 layers x 128x384) into bf16
// WMMA B-fragment order: [[mat][tn][kk][lane][e]] so each lane loads one
// contiguous 32B chunk per (tile, k-step).
// ---------------------------------------------------------------------------
__global__ __launch_bounds__(256)
void swizzle_weights(const float* __restrict__ wx, const float* __restrict__ wh,
                     unsigned short* __restrict__ wsw_us) {
    __bf16* wsw = (__bf16*)wsw_us;
    int idx = blockIdx.x * 256 + threadIdx.x;
    if (idx >= 4 * WMAT) return;
    int mat = idx / WMAT;
    int r   = idx % WMAT;
    int e    = r & 15;
    int lane = (r >> 4) & 31;
    int kk   = (r >> 9) & 3;
    int tn   = r >> 11;                 // 0..23
    int k    = kk * 32 + frag_k(lane, e);
    int col  = tn * 16 + (lane & 15);
    float v;
    if (mat < 2) v = wx[(size_t)(mat * 128 + k) * G3 + col];
    else         v = wh[(size_t)((mat - 2) * 128 + k) * G3 + col];
    wsw[idx] = f2bf(v);
}

// ---------------------------------------------------------------------------
// Fused per-level DAG-GRU step:
//   m  = mean of 8 predecessors' h        (gather, deg == 8 by construction)
//   g  = x_tile @ Wx + b                  (bf16 WMMA, f32 accum)
//   gh = m_tile @ Wh                      (bf16 WMMA, f32 accum)
//   r,z = sigmoid(g+gh); n = tanh(g_n + r*gh_n); h = (1-z)*n + z*m
// One block = 16 nodes x 384 columns. 8 waves x 3 N-tiles x 4 K-steps.
// The lvl==0 / lvl>0 specialization is hoisted out of the WMMA loop so the
// inner loop is pure load+wmma (no per-iteration branches / acc copies).
// ---------------------------------------------------------------------------
__global__ __launch_bounds__(256)
void daggru_level(const float* __restrict__ xin,          // layer input (all nodes)
                  float* __restrict__ hbuf,               // layer output h (all nodes)
                  const unsigned short* __restrict__ wsw_us,
                  const float* __restrict__ biases,
                  const int* __restrict__ edge_src,
                  int layer, int lvl) {
    __shared__ __bf16 lds_xs[2048] __attribute__((aligned(32)));  // x A-frags
    __shared__ __bf16 lds_ms[2048] __attribute__((aligned(32)));  // m A-frags
    __shared__ float  lds_mf[16][HD];                             // m in f32 (for blend)
    __shared__ float  lds_g [16][388];                            // g  (padded stride)
    __shared__ float  lds_gh[16][388];                            // gh

    const __bf16* wsw = (const __bf16*)wsw_us;
    const int tid   = threadIdx.x;
    const int s     = lvl * NPL;
    const int node0 = s + ((int)blockIdx.x << 4);

    // --- 1) gather-mean m into lds_mf (f32) ---
    if (lvl > 0) {
        for (int i = tid; i < 16 * HD; i += 256) {
            int r = i >> 7, c = i & 127;
            int node = node0 + r;
            float acc = 0.f;
            if (node < s + NPL) {
                const int eb = (lvl - 1) * EPL + (node - s) * DEG;
                #pragma unroll
                for (int e = 0; e < DEG; ++e)
                    acc += hbuf[(size_t)edge_src[eb + e] * HD + c];
            }
            lds_mf[r][c] = acc * 0.125f;
        }
    } else {
        for (int i = tid; i < 16 * HD; i += 256) lds_mf[i >> 7][i & 127] = 0.f;
    }

    // --- 2) stage x tile directly into A-fragment order (f32 -> bf16) ---
    for (int i = tid; i < 2048; i += 256) {
        int e    = i & 15;
        int lane = (i >> 4) & 31;
        int kk   = i >> 9;
        int k    = kk * 32 + frag_k(lane, e);
        int m    = lane & 15;
        int node = node0 + m;
        float v = (node < s + NPL) ? xin[(size_t)node * HD + k] : 0.f;
        lds_xs[i] = f2bf(v);
    }
    __syncthreads();

    // --- 3) swizzle m into A-fragment order (only needed when lvl > 0) ---
    if (lvl > 0) {
        for (int i = tid; i < 2048; i += 256) {
            int e    = i & 15;
            int lane = (i >> 4) & 31;
            int kk   = i >> 9;
            int k    = kk * 32 + frag_k(lane, e);
            lds_ms[i] = f2bf(lds_mf[lane & 15][k]);
        }
    }
    __syncthreads();

    // --- 4) dual GEMM: g = x@Wx + b ; gh = m@Wh (f32 accum via WMMA) ---
    const int wave = tid >> 5;
    const int lane = tid & 31;
    const float*  b  = biases + layer * G3;
    const __bf16* wx = wsw + (size_t)layer * WMAT;
    const __bf16* wh = wsw + (size_t)(2 + layer) * WMAT;

    v8f accg[3], acch[3];
    #pragma unroll
    for (int q = 0; q < 3; ++q) {
        int tn = wave * 3 + q;
        float bias = b[tn * 16 + (lane & 15)];
        #pragma unroll
        for (int r = 0; r < 8; ++r) { accg[q][r] = bias; acch[q][r] = 0.f; }
    }

    if (lvl > 0) {
        #pragma unroll
        for (int kk = 0; kk < 4; ++kk) {
            v16bf ax = *(const v16bf*)(lds_xs + ((kk * 32 + lane) << 4));
            v16bf am = *(const v16bf*)(lds_ms + ((kk * 32 + lane) << 4));
            #pragma unroll
            for (int q = 0; q < 3; ++q) {
                int tn = wave * 3 + q;
                v16bf bx = *(const v16bf*)(wx + (((size_t)(tn * 4 + kk) * 32 + lane) << 4));
                v16bf bh = *(const v16bf*)(wh + (((size_t)(tn * 4 + kk) * 32 + lane) << 4));
                accg[q] = __builtin_amdgcn_wmma_f32_16x16x32_bf16(
                    false, ax, false, bx, (short)0, accg[q], false, false);
                acch[q] = __builtin_amdgcn_wmma_f32_16x16x32_bf16(
                    false, am, false, bh, (short)0, acch[q], false, false);
            }
        }
    } else {
        #pragma unroll
        for (int kk = 0; kk < 4; ++kk) {
            v16bf ax = *(const v16bf*)(lds_xs + ((kk * 32 + lane) << 4));
            #pragma unroll
            for (int q = 0; q < 3; ++q) {
                int tn = wave * 3 + q;
                v16bf bx = *(const v16bf*)(wx + (((size_t)(tn * 4 + kk) * 32 + lane) << 4));
                accg[q] = __builtin_amdgcn_wmma_f32_16x16x32_bf16(
                    false, ax, false, bx, (short)0, accg[q], false, false);
            }
        }
    }

    // --- 5) spill accumulators to LDS (C layout: lanes 0-15 -> M=r, 16-31 -> M=8+r)
    #pragma unroll
    for (int q = 0; q < 3; ++q) {
        int tn  = wave * 3 + q;
        int col = tn * 16 + (lane & 15);
        int rb  = (lane >> 4) << 3;
        #pragma unroll
        for (int r = 0; r < 8; ++r) {
            lds_g [rb + r][col] = accg[q][r];
            lds_gh[rb + r][col] = acch[q][r];
        }
    }
    __syncthreads();

    // --- 6) GRU gating + write h ---
    for (int i = tid; i < 16 * HD; i += 256) {
        int r = i >> 7, jc = i & 127;
        int node = node0 + r;
        if (node >= s + NPL) continue;
        float gr = lds_g[r][jc]        + lds_gh[r][jc];
        float gz = lds_g[r][128 + jc]  + lds_gh[r][128 + jc];
        float rv = 1.f / (1.f + __expf(-gr));
        float zv = 1.f / (1.f + __expf(-gz));
        float nv = tanhf(lds_g[r][256 + jc] + rv * lds_gh[r][256 + jc]);
        float mv = lds_mf[r][jc];
        hbuf[(size_t)node * HD + jc] = (1.f - zv) * nv + zv * mv;
    }
}

extern "C" void kernel_launch(void* const* d_in, const int* in_sizes, int n_in,
                              void* d_out, int out_size, void* d_ws, size_t ws_size,
                              hipStream_t stream) {
    const float* feats    = (const float*)d_in[0];
    const float* wx       = (const float*)d_in[1];
    const float* wh       = (const float*)d_in[2];
    const float* biases   = (const float*)d_in[3];
    const int*   edge_src = (const int*)d_in[4];
    float* out = (float*)d_out;

    // workspace: [h1: 200000*128 f32][swizzled bf16 weights: 4 * 49152]
    float* h1 = (float*)d_ws;
    unsigned short* wsw =
        (unsigned short*)((char*)d_ws + (size_t)NTOT * HD * sizeof(float));

    swizzle_weights<<<(4 * WMAT + 255) / 256, 256, 0, stream>>>(wx, wh, wsw);

    const int blocks = (NPL + 15) / 16;  // 782
    for (int layer = 0; layer < 2; ++layer) {
        const float* xin = layer ? h1 : feats;
        float* hb = layer ? out : h1;
        for (int lvl = 0; lvl < NLVL; ++lvl) {
            daggru_level<<<blocks, 256, 0, stream>>>(
                xin, hb, wsw, biases, edge_src, layer, lvl);
        }
    }
}
